// Seq2Seq_LSTM_17566416241136
// MI455X (gfx1250) — matrix operand order repaired and checked by reading, not compiled, not run on hardware
//
#include <hip/hip_runtime.h>
#include <hip/hip_bf16.h>

// ---------------------------------------------------------------------------
// Seq2Seq LSTM on gfx1250 (MI455X).
// bf16 WMMA (16x16x32) GEMMs, f32 accumulate. Fused [h|x] state vector vs
// packed [W_hh|W_ih] weight: each timestep = ONE GEMM (256x1184)@(1184x4096)
// + pointwise gates. Gates GEMM uses 64x64 register blocking per wave
// (16 accumulators) -> 1.0 b128 loads per WMMA, ~4x less L2 traffic than
// a tile-per-wave scheme. Weights bf16-packed once per call (~10 MB, L2-res).
// ---------------------------------------------------------------------------

#define RNN   1024
#define POSE  135
#define SEEDL 120
#define TGT   24
#define BB    256
#define XPAD  160                  // POSE padded to multiple of 32
#define SDIM  (RNN + XPAD)         // 1184, K of the fused GEMM (multiple of 32)
#define NG    (4 * RNN)            // 4096 gate outputs
#define NFC   144                  // POSE padded to multiple of 16 for fc GEMM

typedef __attribute__((ext_vector_type(16))) __bf16 bf16x16;
typedef __attribute__((ext_vector_type(8)))  __bf16 bf16x8;
typedef __attribute__((ext_vector_type(8)))  float  f32x8;

union BfPack { bf16x16 v16; bf16x8 v8[2]; };

// ---------------------------------------------------------------------------
// Gates GEMM, specialized: gates[256 x 4096] = S[256 x SDIM] @ Wc[4096 x SDIM]^T
// Each wave computes a 64x64 block: 4 A fragments x 4 B fragments -> 16 WMMAs
// per K-step, 16 b128 loads per K-step (1.0 loads/WMMA).
// Grid: 4 m-blocks * 64 n-blocks = 256 waves = 32 blocks of 256 threads.
// ---------------------------------------------------------------------------
__global__ __launch_bounds__(256)
void lstm_gates_gemm(const __bf16* __restrict__ S,
                     const __bf16* __restrict__ Wc,
                     float* __restrict__ gates) {
  const int wave = (blockIdx.x * blockDim.x + threadIdx.x) >> 5; // 0..255
  const int lane = threadIdx.x & 31;
  const int hi   = lane >> 4;
  const int l16  = lane & 15;
  const int bm   = (wave >> 6) * 64;   // 4 m-blocks over M=256
  const int bn   = (wave & 63) * 64;   // 64 n-blocks over N=4096

  // A layout (16x32 bf16): lane = M row; VGPR0..3 K=hi*8.., VGPR4..7 K=16+hi*8..
  const __bf16* arow = S  + (size_t)(bm + l16) * SDIM;
  // B layout (32x16 bf16): lane%16 = N col; VGPR0..3 K=hi*8.., VGPR4..7 K=16+hi*8..
  const __bf16* brow = Wc + (size_t)(bn + l16) * SDIM;

  f32x8 acc[4][4] = {};
  for (int k = 0; k < SDIM; k += 32) {
    BfPack a[4], b[4];
#pragma unroll
    for (int mi = 0; mi < 4; ++mi) {
      const __bf16* ap = arow + (size_t)mi * 16 * SDIM + k;
      a[mi].v8[0] = *(const bf16x8*)(ap + hi * 8);
      a[mi].v8[1] = *(const bf16x8*)(ap + 16 + hi * 8);
    }
#pragma unroll
    for (int ni = 0; ni < 4; ++ni) {
      const __bf16* bp = brow + (size_t)ni * 16 * SDIM + k + hi * 8;
      b[ni].v8[0] = *(const bf16x8*)(bp);
      b[ni].v8[1] = *(const bf16x8*)(bp + 16);
    }
#pragma unroll
    for (int mi = 0; mi < 4; ++mi)
#pragma unroll
      for (int ni = 0; ni < 4; ++ni)
        acc[mi][ni] = __builtin_amdgcn_wmma_f32_16x16x32_bf16(
            false, a[mi].v16, false, b[ni].v16, (short)0, acc[mi][ni],
            false, false);
  }

  // D layout: VGPR r -> row = r + 8*hi, col = lane%16.
#pragma unroll
  for (int mi = 0; mi < 4; ++mi) {
    float* crow = gates + (size_t)(bm + mi * 16 + hi * 8) * NG + bn + l16;
#pragma unroll
    for (int ni = 0; ni < 4; ++ni)
#pragma unroll
      for (int r = 0; r < 8; ++r)
        crow[(size_t)r * NG + ni * 16] = acc[mi][ni][r];
  }
}

// ---------------------------------------------------------------------------
// Generic GEMM (one 16x16 tile per wave) for the small fc matmul:
// C[M x N] (f32) = A[M x K] (bf16, lda) @ B[N x K]^T (bf16, ldb).
// ---------------------------------------------------------------------------
__global__ void gemm_bf16_wmma(const __bf16* __restrict__ A, int lda,
                               const __bf16* __restrict__ B, int ldb,
                               float* __restrict__ C, int ldc,
                               int ntilesN, int K) {
  const int wave = (blockIdx.x * blockDim.x + threadIdx.x) >> 5;
  const int lane = threadIdx.x & 31;
  const int tm   = wave / ntilesN;
  const int tn   = wave % ntilesN;
  const int hi   = lane >> 4;
  const int l16  = lane & 15;

  const __bf16* arow = A + (size_t)(tm * 16 + l16) * lda;
  const __bf16* brow = B + (size_t)(tn * 16 + l16) * ldb;

  f32x8 acc = {};
  for (int k = 0; k < K; k += 32) {
    BfPack a, b;
    a.v8[0] = *(const bf16x8*)(arow + k + hi * 8);
    a.v8[1] = *(const bf16x8*)(arow + k + 16 + hi * 8);
    b.v8[0] = *(const bf16x8*)(brow + k + hi * 8);
    b.v8[1] = *(const bf16x8*)(brow + k + 16 + hi * 8);
    acc = __builtin_amdgcn_wmma_f32_16x16x32_bf16(
        false, a.v16, false, b.v16, (short)0, acc, false, false);
  }
  float* crow = C + (size_t)(tm * 16 + hi * 8) * ldc + tn * 16 + l16;
#pragma unroll
  for (int r = 0; r < 8; ++r) crow[(size_t)r * ldc] = acc[r];
}

// ---------------------------------------------------------------------------
// Pointwise LSTM cell update. idx = b*RNN + j.
// gates row layout: [i(1024) | f(1024) | g(1024) | o(1024)].
// ---------------------------------------------------------------------------
__global__ void lstm_pointwise(const float* __restrict__ gates,
                               float* __restrict__ c,
                               __bf16* __restrict__ S) {
  const int idx = blockIdx.x * blockDim.x + threadIdx.x;
  const int b = idx >> 10, j = idx & (RNN - 1);
  const float* g = gates + (size_t)b * NG;
  const float i_ = g[j];
  const float f_ = g[j + RNN];
  const float g_ = g[j + 2 * RNN];
  const float o_ = g[j + 3 * RNN];
  const float si = 1.f / (1.f + __expf(-i_));
  const float sf = 1.f / (1.f + __expf(-f_));
  const float so = 1.f / (1.f + __expf(-o_));
  const float cn = sf * c[idx] + si * tanhf(g_);
  const float hn = so * tanhf(cn);
  c[idx] = cn;
  S[(size_t)b * SDIM + j] = (__bf16)hn;
}

// Feed next encoder input (poses[:, col, :]) into the x-slot of S; optionally
// also capture it as the f32 decoder input buffer.
__global__ void set_x(const float* __restrict__ poses, int col,
                      __bf16* __restrict__ S, float* __restrict__ inp_opt) {
  const int idx = blockIdx.x * blockDim.x + threadIdx.x;
  if (idx >= BB * POSE) return;
  const int b = idx / POSE, p = idx % POSE;
  const float v = poses[((size_t)b * (SEEDL + TGT) + col) * POSE + p];
  S[(size_t)b * SDIM + RNN + p] = (__bf16)v;
  if (inp_opt) inp_opt[idx] = v;
}

// Decoder output: out = inp + fc + b_fc; feed back as next x (bf16) and inp.
__global__ void dec_out(const float* __restrict__ fc,
                        const float* __restrict__ bfc,
                        float* __restrict__ inp,
                        __bf16* __restrict__ S,
                        float* __restrict__ out, int d) {
  const int idx = blockIdx.x * blockDim.x + threadIdx.x;
  if (idx >= BB * POSE) return;
  const int b = idx / POSE, p = idx % POSE;
  const float v = inp[idx] + fc[(size_t)b * NFC + p] + bfc[p];
  out[((size_t)b * TGT + d) * POSE + p] = v;
  S[(size_t)b * SDIM + RNN + p] = (__bf16)v;
  inp[idx] = v;
}

// Pack fused bf16 weight Wc[4096 x 1184] = [W_hh | W_ih | 0-pad].
__global__ void pack_wc(const float* __restrict__ Wih,
                        const float* __restrict__ Whh,
                        __bf16* __restrict__ Wc) {
  const long idx = (long)blockIdx.x * blockDim.x + threadIdx.x;
  if (idx >= (long)NG * SDIM) return;
  const int n = (int)(idx / SDIM), k = (int)(idx % SDIM);
  float v;
  if (k < RNN)                 v = Whh[(size_t)n * RNN + k];
  else if (k - RNN < POSE)     v = Wih[(size_t)n * POSE + (k - RNN)];
  else                         v = 0.f;
  Wc[idx] = (__bf16)v;
}

// Pack bf16 Wfc[144 x 1024] (rows >= 135 zero).
__global__ void pack_wfc(const float* __restrict__ Wfc_f32,
                         __bf16* __restrict__ Wfc) {
  const long idx = (long)blockIdx.x * blockDim.x + threadIdx.x;
  if (idx >= (long)NFC * RNN) return;
  const int p = (int)(idx / RNN), k = (int)(idx % RNN);
  Wfc[idx] = (__bf16)((p < POSE) ? Wfc_f32[(size_t)p * RNN + k] : 0.f);
}

__global__ void zero_bf16(__bf16* p, long n) {
  const long i = (long)blockIdx.x * blockDim.x + threadIdx.x;
  if (i < n) p[i] = (__bf16)0.f;
}
__global__ void zero_f32(float* p, long n) {
  const long i = (long)blockIdx.x * blockDim.x + threadIdx.x;
  if (i < n) p[i] = 0.f;
}

// ---------------------------------------------------------------------------
extern "C" void kernel_launch(void* const* d_in, const int* in_sizes, int n_in,
                              void* d_out, int out_size, void* d_ws, size_t ws_size,
                              hipStream_t stream) {
  const float* poses = (const float*)d_in[0];  // (256,144,135)
  const float* W_ih  = (const float*)d_in[1];  // (4096,135)
  const float* W_hh  = (const float*)d_in[2];  // (4096,1024)
  const float* W_fc  = (const float*)d_in[3];  // (135,1024)
  const float* b_fc  = (const float*)d_in[4];  // (135,)
  float* out = (float*)d_out;                  // (256,24,135)

  // Workspace carve-up (256B aligned).
  char* ws = (char*)d_ws;
  auto align = [](size_t x) { return (x + 255) & ~(size_t)255; };
  __bf16* Wc   = (__bf16*)ws;  ws += align((size_t)NG * SDIM * 2);    // 9.70 MB
  __bf16* Wfc  = (__bf16*)ws;  ws += align((size_t)NFC * RNN * 2);    // 0.29 MB
  __bf16* S    = (__bf16*)ws;  ws += align((size_t)BB * SDIM * 2);    // 0.61 MB
  float* gates = (float*)ws;   ws += align((size_t)BB * NG * 4);      // 4.19 MB
  float* cbuf  = (float*)ws;   ws += align((size_t)BB * RNN * 4);     // 1.05 MB
  float* inp   = (float*)ws;   ws += align((size_t)BB * POSE * 4);    // 0.14 MB
  float* fcbuf = (float*)ws;   ws += align((size_t)BB * NFC * 4);     // 0.15 MB
  (void)ws_size; (void)n_in; (void)in_sizes; (void)out_size;

  const int T = 256;

  // --- One-time (per call) prep: bf16 weight packing, state init ------------
  {
    long n = (long)NG * SDIM;
    pack_wc<<<(int)((n + T - 1) / T), T, 0, stream>>>(W_ih, W_hh, Wc);
    n = (long)NFC * RNN;
    pack_wfc<<<(int)((n + T - 1) / T), T, 0, stream>>>(W_fc, Wfc);
    n = (long)BB * SDIM;
    zero_bf16<<<(int)((n + T - 1) / T), T, 0, stream>>>(S, n);   // h=0, x pad=0
    n = (long)BB * RNN;
    zero_f32<<<(int)((n + T - 1) / T), T, 0, stream>>>(cbuf, n); // c=0
    set_x<<<(BB * POSE + T - 1) / T, T, 0, stream>>>(poses, 0, S, nullptr);
  }

  const int gatesBlocks = (4 * 64) / 8;           // 256 waves, 8 per block = 32
  const int fcTiles     = (BB / 16) * (NFC / 16); // 144 tiles -> 18 blocks

  // --- Encoder: 119 steps ---------------------------------------------------
  for (int t = 0; t < SEEDL - 1; ++t) {
    lstm_gates_gemm<<<gatesBlocks, T, 0, stream>>>(S, Wc, gates);
    lstm_pointwise<<<(BB * RNN) / T, T, 0, stream>>>(gates, cbuf, S);
    const bool last = (t == SEEDL - 2);
    set_x<<<(BB * POSE + T - 1) / T, T, 0, stream>>>(
        poses, t + 1, S, last ? inp : nullptr);
  }

  // --- Decoder: 24 steps ----------------------------------------------------
  for (int d = 0; d < TGT; ++d) {
    lstm_gates_gemm<<<gatesBlocks, T, 0, stream>>>(S, Wc, gates);
    lstm_pointwise<<<(BB * RNN) / T, T, 0, stream>>>(gates, cbuf, S);
    // fc = h @ W_fc^T : A is the h-slice of S (lda = SDIM, K = RNN).
    gemm_bf16_wmma<<<fcTiles / 8, T, 0, stream>>>(
        S, SDIM, Wfc, RNN, fcbuf, NFC, NFC / 16, RNN);
    dec_out<<<(BB * POSE + T - 1) / T, T, 0, stream>>>(
        fcbuf, b_fc, inp, S, out, d);
  }
}
